// MultiHeadAttention_36301063586232
// MI455X (gfx1250) — matrix operand order repaired and checked by reading, not compiled, not run on hardware
//
#include <hip/hip_runtime.h>
#include <stdint.h>

// ---------------- problem constants ----------------
#define Bn   4
#define Sn   2048
#define Dn   1024
#define Hn   8
#define DHn  128
#define MTOT (Bn * Sn)   // 8192 rows

typedef __attribute__((ext_vector_type(16))) __bf16 v16bf;
typedef __attribute__((ext_vector_type(8)))  float  v8f;
typedef __attribute__((ext_vector_type(4)))  int    v4i;

#define GLAS  __attribute__((address_space(1)))
#define LDSAS __attribute__((address_space(3)))

#if __has_builtin(__builtin_amdgcn_global_load_async_to_lds_b128) && \
    __has_builtin(__builtin_amdgcn_s_wait_asynccnt)
#define HAVE_ASYNC_LDS 1
#else
#define HAVE_ASYNC_LDS 0
#endif

union FragU {
    v16bf          v;
    uint4          q[2];
    unsigned short h[16];
};

// round-to-nearest-even f32 -> raw bf16 bits
__device__ __forceinline__ unsigned short f2bf(float f) {
    unsigned int u = __float_as_uint(f);
    u += 0x7fffu + ((u >> 16) & 1u);
    return (unsigned short)(u >> 16);
}

// A-fragment (16x32, 16-bit): lane l holds row m=l%16.
// lanes 0-15: elems0-15 = K k1..k1+15 (k1=kc)
// lanes16-31: same with k1 = kc+16  (contiguous, matches B)
__device__ __forceinline__ v16bf load_a(const unsigned short* rowbase, int k1) {
    FragU f;
    f.q[0] = *(const uint4*)(rowbase + k1);
    f.q[1] = *(const uint4*)(rowbase + k1 + 8);
    return f.v;
}

// B-fragment (32x16, 16-bit): lane l holds col n=l%16;
// lanes 0-15 hold K=kc..kc+15, lanes 16-31 hold K=kc+16..kc+31 (contiguous).
__device__ __forceinline__ v16bf load_b(const unsigned short* rowbase, int k0) {
    FragU f;
    f.q[0] = *(const uint4*)(rowbase + k0);
    f.q[1] = *(const uint4*)(rowbase + k0 + 8);
    return f.v;
}

__device__ __forceinline__ v8f wmma_bf16(v16bf a, v16bf b, v8f c) {
    return __builtin_amdgcn_wmma_f32_16x16x32_bf16(false, a, false, b, (short)0, c,
                                                   false, false);
}

// 16-byte global -> LDS copy: async (ASYNCcnt-tracked) when available.
__device__ __forceinline__ void cp16_g2l(const unsigned short* g, unsigned short* l) {
#if HAVE_ASYNC_LDS
    __builtin_amdgcn_global_load_async_to_lds_b128(
        (GLAS v4i*)(uintptr_t)g, (LDSAS v4i*)l, 0, 0);
#else
    *(uint4*)l = *(const uint4*)g;
#endif
}

__device__ __forceinline__ void wait_stage() {
#if HAVE_ASYNC_LDS
    __builtin_amdgcn_s_wait_asynccnt(0);
#endif
    __syncthreads();
}

// ---------------- kernel 1: f32 -> bf16 cast ----------------
__global__ __launch_bounds__(256) void cast_bf16(const float* __restrict__ in,
                                                 unsigned short* __restrict__ out, int n) {
    int i = blockIdx.x * 256 + threadIdx.x;
    if (i < n) out[i] = f2bf(in[i]);
}

// ---------------- kernel 2: WMMA GEMM  P = X * W^T ----------------
// X: (Mtot x 1024) bf16 row-major.  W: (1024 x 1024) bf16 row-major (n,k) => B operand.
// vmode==0: out[m*D + n].   vmode==1 (V): out = Vt[((n>>7)*4 + (m>>11))*128 + (n&127)][m&2047]
__global__ __launch_bounds__(256) void gemm_bf16(const unsigned short* __restrict__ X,
                                                 const unsigned short* __restrict__ W,
                                                 unsigned short* __restrict__ out,
                                                 float outScale, int vmode) {
    const int lane = threadIdx.x & 31;
    const int wid  = threadIdx.x >> 5;         // 0..7
    const int lr   = lane & 15;
    const int hi   = lane >> 4;                // 0 or 1
    const int waveM = (wid & 3) * 32;          // 4 waves down M  -> 128
    const int waveN = (wid >> 2) * 64;         // 2 waves across N -> 128
    const int blockM = blockIdx.x * 128;
    const int blockN = blockIdx.y * 128;

    v8f acc[2][4];
#pragma unroll
    for (int mt = 0; mt < 2; ++mt)
#pragma unroll
        for (int nt = 0; nt < 4; ++nt) acc[mt][nt] = (v8f){};

    const unsigned short* a0 = X + (size_t)(blockM + waveM + lr) * Dn;
    const unsigned short* a1 = a0 + (size_t)16 * Dn;
    const unsigned short* bp[4];
#pragma unroll
    for (int nt = 0; nt < 4; ++nt)
        bp[nt] = W + (size_t)(blockN + waveN + nt * 16 + lr) * Dn;

    for (int kc = 0; kc < Dn; kc += 32) {
        // prefetch the streaming A operand a couple of k-steps ahead (global_prefetch_b8)
        __builtin_prefetch((const void*)(a0 + kc + 128), 0, 0);
        __builtin_prefetch((const void*)(a1 + kc + 128), 0, 0);

        const int k1 = kc + (hi ? 16 : 0);
        v16bf af0 = load_a(a0, k1);
        v16bf af1 = load_a(a1, k1);
        const int k0 = kc + (hi ? 16 : 0);
#pragma unroll
        for (int nt = 0; nt < 4; ++nt) {
            v16bf bfm = load_b(bp[nt], k0);
            acc[0][nt] = wmma_bf16(af0, bfm, acc[0][nt]);
            acc[1][nt] = wmma_bf16(af1, bfm, acc[1][nt]);
        }
    }

    // C/D layout: VGPR r -> row r (lanes 0-15) / r+8 (lanes 16-31); col = lane%16
#pragma unroll
    for (int mt = 0; mt < 2; ++mt)
#pragma unroll
        for (int nt = 0; nt < 4; ++nt)
#pragma unroll
            for (int r = 0; r < 8; ++r) {
                int m = blockM + waveM + mt * 16 + r + 8 * hi;
                int n = blockN + waveN + nt * 16 + lr;
                unsigned short bv = f2bf(acc[mt][nt][r] * outScale);
                if (vmode == 0) {
                    out[(size_t)m * Dn + n] = bv;
                } else {
                    int h = n >> 7, j = n & 127, bb = m >> 11, s = m & 2047;
                    out[((size_t)((h * Bn + bb) * DHn + j) << 11) + s] = bv;
                }
            }
}

// ---------------- kernel 3: flash attention ----------------
// One wave = 16 queries; block = 4 waves sharing async-staged K/V tiles in LDS.
// Qp,Kp: (B*S x D) bf16 (Qp pre-scaled by 1/sqrt(D)).  Vt: per (h,b) transposed (128 x 2048) bf16.
// xout: f32 attention output + residual query, (B*S x D).
__global__ __launch_bounds__(128) void attn_fa(const unsigned short* __restrict__ Qp,
                                               const unsigned short* __restrict__ Kp,
                                               const unsigned short* __restrict__ Vt,
                                               const float* __restrict__ query,
                                               float* __restrict__ xout) {
    __shared__ unsigned short ktile[2][32][128];  // [buf][key][dh]   16 KB
    __shared__ unsigned short vtile[2][128][32];  // [buf][j][key]    16 KB
    __shared__ float plds[4][16][33];             // per-wave P staging

    const int tid  = threadIdx.x;
    const int lane = tid & 31;
    const int w    = tid >> 5;               // wave in block, 0..3
    const int lr   = lane & 15;
    const int hi   = lane >> 4;
    const int hb   = blockIdx.y;             // 0..31
    const int h    = hb >> 2;                // head
    const int b    = hb & 3;                 // batch
    const int sq   = blockIdx.x * 64 + w * 16;

    const unsigned short* kbase = Kp + (size_t)(b * Sn) * Dn + h * DHn;
    const unsigned short* vbase = Vt + (size_t)((h * Bn + b) * DHn) * Sn;

    // stage one 32-key K tile (32x128) + V tile (128x32) into LDS buffer bf_
    auto stage = [&](int kb_, int bf_) {
#pragma unroll
        for (int i = 0; i < 4; ++i) {                 // K: 512 x 16B chunks / 128 thr
            int c = i * 128 + tid;
            int key = c >> 4, eo = (c & 15) * 8;
            cp16_g2l(kbase + (size_t)(kb_ + key) * Dn + eo, &ktile[bf_][key][eo]);
        }
#pragma unroll
        for (int i = 0; i < 4; ++i) {                 // V: 512 x 16B chunks / 128 thr
            int c = i * 128 + tid;
            int j = c >> 2, eo = (c & 3) * 8;
            cp16_g2l(vbase + (size_t)j * Sn + kb_ + eo, &vtile[bf_][j][eo]);
        }
    };

    // resident Q fragments over DH=128 (4 k-chunks of 32)
    const unsigned short* qrow = Qp + (size_t)(b * Sn + sq + lr) * Dn + h * DHn;
    v16bf qa[4];
#pragma unroll
    for (int c = 0; c < 4; ++c) qa[c] = load_a(qrow, c * 32 + (hi ? 16 : 0));

    v8f o[8];
#pragma unroll
    for (int t = 0; t < 8; ++t) o[t] = (v8f){};
    float rmax[8], rsum[8];
#pragma unroll
    for (int r = 0; r < 8; ++r) { rmax[r] = -INFINITY; rsum[r] = 0.f; }

    stage(0, 0);
    int buf = 0;

    for (int kb = 0; kb < Sn; kb += 32) {
        wait_stage();                       // own async copies done + all waves arrived
        if (kb + 32 < Sn) stage(kb + 32, buf ^ 1);

        // ---- scores: two 16x16 C-tiles over 32 keys, K-dim = DH = 128 (from LDS) ----
        v8f s0 = (v8f){}, s1 = (v8f){};
        const unsigned short* k0p = &ktile[buf][lr][0];
        const unsigned short* k1p = &ktile[buf][16 + lr][0];
#pragma unroll
        for (int c = 0; c < 4; ++c) {
            const int kk = c * 32 + (hi ? 16 : 0);
            s0 = wmma_bf16(qa[c], load_b(k0p, kk), s0);
            s1 = wmma_bf16(qa[c], load_b(k1p, kk), s1);
        }

        // ---- online softmax (rows live across 16-lane halves) ----
        float pf[8];
#pragma unroll
        for (int r = 0; r < 8; ++r) {
            float bm = fmaxf(s0[r], s1[r]);
#pragma unroll
            for (int off = 1; off < 16; off <<= 1)
                bm = fmaxf(bm, __shfl_xor(bm, off, 32));
            float nm = fmaxf(rmax[r], bm);
            float p0 = __expf(s0[r] - nm);
            float p1 = __expf(s1[r] - nm);
            float ls = p0 + p1;
#pragma unroll
            for (int off = 1; off < 16; off <<= 1)
                ls += __shfl_xor(ls, off, 32);
            float fac = __expf(rmax[r] - nm);
            rsum[r] = rsum[r] * fac + ls;
            rmax[r] = nm;
            pf[r]   = fac;
            plds[w][r + 8 * hi][lr]      = p0;   // P in C-layout -> LDS
            plds[w][r + 8 * hi][16 + lr] = p1;
        }
#pragma unroll
        for (int t = 0; t < 8; ++t)
#pragma unroll
            for (int r = 0; r < 8; ++r) o[t][r] *= pf[r];

        // ---- re-load P as A-fragment (16x32) from LDS, cvt to bf16 ----
        FragU pa;
#pragma unroll
        for (int e = 0; e < 8; ++e) {
            const int klo = (hi ? 16 : 0) + e;
            pa.h[e]     = f2bf(plds[w][lr][klo]);
            pa.h[e + 8] = f2bf(plds[w][lr][klo + 8]);
        }

        // ---- O += P (16x32) x V (32x16) for 8 j-tiles (V from LDS) ----
#pragma unroll
        for (int t = 0; t < 8; ++t) {
            const unsigned short* vp = &vtile[buf][t * 16 + lr][hi ? 16 : 0];
            FragU vf;
            vf.q[0] = *(const uint4*)vp;
            vf.q[1] = *(const uint4*)(vp + 8);
            o[t] = wmma_bf16(pa.v, vf.v, o[t]);
        }
        buf ^= 1;
    }

    // ---- epilogue: normalize rows, add residual, store f32 ----
#pragma unroll
    for (int r = 0; r < 8; ++r) {
        const int m = b * Sn + sq + r + 8 * hi;
        const float inv = 1.0f / rsum[r];
#pragma unroll
        for (int t = 0; t < 8; ++t) {
            size_t idx = (size_t)m * Dn + h * DHn + t * 16 + lr;
            xout[idx] = o[t][r] * inv + query[idx];
        }
    }
}

// ---------------- kernel 4: BN stats (deterministic, block per feature) ----------------
__global__ __launch_bounds__(256) void bn_stats(const float* __restrict__ x,
                                                float* __restrict__ meanp,
                                                float* __restrict__ rstdp) {
    __shared__ float sh[256], sh2[256];
    const int f = blockIdx.x;
    const int t = threadIdx.x;
    float s = 0.f, ss = 0.f;
    for (int m = t; m < MTOT; m += 256) {
        float v = x[(size_t)m * Dn + f];
        s += v; ss += v * v;
    }
    sh[t] = s; sh2[t] = ss;
    __syncthreads();
    for (int o = 128; o > 0; o >>= 1) {
        if (t < o) { sh[t] += sh[t + o]; sh2[t] += sh2[t + o]; }
        __syncthreads();
    }
    if (t == 0) {
        float mu  = sh[0] / (float)MTOT;
        float var = sh2[0] / (float)MTOT - mu * mu;
        meanp[f] = mu;
        rstdp[f] = rsqrtf(var + 1e-5f);
    }
}

// ---------------- kernel 5: BN apply ----------------
__global__ __launch_bounds__(256) void bn_apply(const float* __restrict__ x,
                                                const float* __restrict__ meanp,
                                                const float* __restrict__ rstdp,
                                                const float* __restrict__ gamma,
                                                const float* __restrict__ beta,
                                                float* __restrict__ out) {
    size_t i = (size_t)blockIdx.x * 256 + threadIdx.x;
    int f = (int)(i & (Dn - 1));
    out[i] = gamma[f] * (x[i] - meanp[f]) * rstdp[f] + beta[f];
}

// ---------------- launcher ----------------
extern "C" void kernel_launch(void* const* d_in, const int* in_sizes, int n_in,
                              void* d_out, int out_size, void* d_ws, size_t ws_size,
                              hipStream_t stream) {
    (void)in_sizes; (void)n_in; (void)out_size; (void)ws_size;
    const float* query = (const float*)d_in[0];
    const float* keys  = (const float*)d_in[1];
    const float* Wq    = (const float*)d_in[2];
    const float* Wk    = (const float*)d_in[3];
    const float* Wv    = (const float*)d_in[4];
    const float* gamma = (const float*)d_in[5];
    const float* beta  = (const float*)d_in[6];

    char* ws = (char*)d_ws;
    const size_t MB = (size_t)1 << 20;
    unsigned short* Xq  = (unsigned short*)(ws + 0 * MB);    // 16 MB
    unsigned short* Xk  = (unsigned short*)(ws + 16 * MB);   // 16 MB
    unsigned short* Wqb = (unsigned short*)(ws + 32 * MB);   // 2 MB
    unsigned short* Wkb = (unsigned short*)(ws + 34 * MB);   // 2 MB
    unsigned short* Wvb = (unsigned short*)(ws + 36 * MB);   // 2 MB
    unsigned short* Qp  = (unsigned short*)(ws + 38 * MB);   // 16 MB
    unsigned short* Kp  = (unsigned short*)(ws + 54 * MB);   // 16 MB
    unsigned short* Vt  = (unsigned short*)(ws + 70 * MB);   // 16 MB
    float*          Xr  = (float*)(ws + 86 * MB);            // 32 MB
    float*          mep = (float*)(ws + 118 * MB);
    float*          rsp = (float*)(ws + 118 * MB + 4096);

    const int NE = MTOT * Dn;   // 8,388,608
    const int NW = Dn * Dn;     // 1,048,576

    cast_bf16<<<NE / 256, 256, 0, stream>>>(query, Xq, NE);
    cast_bf16<<<NE / 256, 256, 0, stream>>>(keys,  Xk, NE);
    cast_bf16<<<NW / 256, 256, 0, stream>>>(Wq, Wqb, NW);
    cast_bf16<<<NW / 256, 256, 0, stream>>>(Wk, Wkb, NW);
    cast_bf16<<<NW / 256, 256, 0, stream>>>(Wv, Wvb, NW);

    dim3 ggrid(MTOT / 128, Dn / 128);   // 64 x 8
    gemm_bf16<<<ggrid, 256, 0, stream>>>(Xq, Wqb, Qp, 0.03125f, 0);  // 1/sqrt(D) folded into Q
    gemm_bf16<<<ggrid, 256, 0, stream>>>(Xk, Wkb, Kp, 1.0f, 0);
    gemm_bf16<<<ggrid, 256, 0, stream>>>(Xk, Wvb, Vt, 1.0f, 1);     // V stored transposed

    attn_fa<<<dim3(Sn / 64, Hn * Bn), 128, 0, stream>>>(Qp, Kp, Vt, query, Xr);

    bn_stats<<<Dn, 256, 0, stream>>>(Xr, mep, rsp);
    bn_apply<<<NE / 256, 256, 0, stream>>>(Xr, mep, rsp, gamma, beta, (float*)d_out);
}
